// Quantizer_58935541236410
// MI455X (gfx1250) — matrix operand order repaired and checked by reading, not compiled, not run on hardware
//
#include <hip/hip_runtime.h>
#include <hip/hip_bf16.h>
#include <math.h>

// ---------------------------------------------------------------------------
// Problem constants (reference: D=256, K=8192, B=8, T=2048 -> N=16384)
// ---------------------------------------------------------------------------
constexpr int Dq   = 256;
constexpr int Kq   = 8192;
constexpr int Nq   = 16384;
constexpr int DP   = 768;          // 3*D : [xh | xl | xh] . [eh | eh | el]
constexpr int LDA  = 776;          // padded LDS row stride in bf16 (1552 B -> bank offset 4)

typedef __attribute__((ext_vector_type(16))) __bf16 v16bf;
typedef __attribute__((ext_vector_type(8)))  float  v8f;
typedef __attribute__((ext_vector_type(4)))  unsigned int u32x4;   // POD 16-byte vector

union ABFrag { u32x4 u[2]; v16bf v; };   // all-POD members -> usable union

// ---------------------------------------------------------------------------
// Prep: split x (fp32) into bf16 hi/lo, layout X2[n][0:256]=xh [256:512]=xl [512:768]=xh
// ---------------------------------------------------------------------------
__global__ __launch_bounds__(256)
void pack_x_kernel(const float* __restrict__ x, __bf16* __restrict__ X2) {
    int idx = blockIdx.x * 256 + threadIdx.x;
    if (idx >= Nq * Dq) return;
    int n = idx >> 8, d = idx & 255;
    float v = x[idx];
    __bf16 h = (__bf16)v;
    __bf16 l = (__bf16)(v - (float)h);
    __bf16* row = X2 + (size_t)n * DP;
    row[d] = h; row[256 + d] = l; row[512 + d] = h;
}

// Prep: E2t[k][0:256]=eh [256:512]=eh [512:768]=el  (code-major for contiguous B frags)
__global__ __launch_bounds__(256)
void pack_e_kernel(const float* __restrict__ e, __bf16* __restrict__ E2t) {
    int k = blockIdx.x * 256 + threadIdx.x;   // coalesced read over k
    int d = blockIdx.y;
    float v = e[(size_t)d * Kq + k];
    __bf16 h = (__bf16)v;
    __bf16 l = (__bf16)(v - (float)h);
    __bf16* row = E2t + (size_t)k * DP;
    row[d] = h; row[256 + d] = h; row[512 + d] = l;
}

// Prep: ||e_k||^2 in fp32 (exact)
__global__ __launch_bounds__(256)
void enorm_kernel(const float* __restrict__ e, float* __restrict__ enorm) {
    int k = blockIdx.x * 256 + threadIdx.x;
    float s = 0.f;
    for (int d = 0; d < Dq; ++d) { float v = e[(size_t)d * Kq + k]; s += v * v; }
    enorm[k] = s;
}

// ---------------------------------------------------------------------------
// Main: distance GEMM + running argmin, 2x2 register blocking per wave.
// Grid: Nq/64 blocks of 256 threads (8 waves).  WG owns 64 rows.
// Wave (w&1) = 32-row half, (w>>1) = column-chunk parity (0..3).
// Per chunk: 32 codes x 32 rows, 24 k-steps x 4 wmma_f32_16x16x32_bf16.
// Each A frag feeds 2 WMMAs, each B frag feeds 2 WMMAs -> 2 loads / WMMA.
// ---------------------------------------------------------------------------
__global__ __launch_bounds__(256)
void vq_argmin_kernel(const __bf16* __restrict__ X2,
                      const __bf16* __restrict__ E2t,
                      const float*  __restrict__ enorm,
                      int*          __restrict__ enc_idx) {
    __shared__ __align__(16) __bf16 sA[64 * LDA];           // ~97 KB
    __shared__ unsigned long long  sBest[64];

    const int tid  = threadIdx.x;
    const int lane = tid & 31;
    const int w    = tid >> 5;
    const int rowBase = blockIdx.x * 64;

    if (tid < 64) sBest[tid] = 0xFFFFFFFFFFFFFFFFull;

    // Cooperative copy A block (64 x 768 bf16) into padded LDS (97 u32x4 per row)
    {
        const u32x4* src = reinterpret_cast<const u32x4*>(X2 + (size_t)rowBase * DP);
        u32x4* dst = reinterpret_cast<u32x4*>(sA);
        for (int i = tid; i < 64 * (DP * 2 / 16); i += 256) {
            int r = i / 96, j = i - r * 96;
            dst[r * 97 + j] = src[i];
        }
    }
    __syncthreads();

    const int rhalf = w & 1;           // 32-row half of the block
    const int par   = w >> 1;          // column-chunk parity 0..3
    const int m     = lane & 15;       // A row in tile / B column in chunk
    const int hi    = lane >> 4;       // lane half selects K sub-ranges
    const __bf16* aBase0 = sA + (rhalf * 32 + m) * LDA;        // row tile 0
    const __bf16* aBase1 = aBase0 + 16 * LDA;                  // row tile 1

    float bestVal[2][8];
    int   bestIdx[2][8];
#pragma unroll
    for (int t = 0; t < 2; ++t)
#pragma unroll
        for (int i = 0; i < 8; ++i) { bestVal[t][i] = 3.0e38f; bestIdx[t][i] = 0; }

    for (int c = par; c < Kq / 32; c += 4) {
        const int kc = c * 32;
        const __bf16* bp0 = E2t + (size_t)(kc + m) * DP + hi * 8;        // codes kc..kc+15
        const __bf16* bp1 = bp0 + (size_t)16 * DP;                        // codes kc+16..kc+31
        v8f acc00 = {0.f,0.f,0.f,0.f,0.f,0.f,0.f,0.f};
        v8f acc01 = acc00, acc10 = acc00, acc11 = acc00;
#pragma unroll 2
        for (int s = 0; s < DP / 32; ++s) {
            const int db = s * 32;
            ABFrag a0, a1, b0, b1;
            a0.u[0] = *reinterpret_cast<const u32x4*>(aBase0 + db + hi * 8);
            a0.u[1] = *reinterpret_cast<const u32x4*>(aBase0 + db + 16 + hi * 8);
            a1.u[0] = *reinterpret_cast<const u32x4*>(aBase1 + db + hi * 8);
            a1.u[1] = *reinterpret_cast<const u32x4*>(aBase1 + db + 16 + hi * 8);
            b0.u[0] = *reinterpret_cast<const u32x4*>(bp0 + db);
            b0.u[1] = *reinterpret_cast<const u32x4*>(bp0 + db + 16);
            b1.u[0] = *reinterpret_cast<const u32x4*>(bp1 + db);
            b1.u[1] = *reinterpret_cast<const u32x4*>(bp1 + db + 16);
            acc00 = __builtin_amdgcn_wmma_f32_16x16x32_bf16(
                        false, a0.v, false, b0.v, (short)0, acc00, false, false);
            acc01 = __builtin_amdgcn_wmma_f32_16x16x32_bf16(
                        false, a0.v, false, b1.v, (short)0, acc01, false, false);
            acc10 = __builtin_amdgcn_wmma_f32_16x16x32_bf16(
                        false, a1.v, false, b0.v, (short)0, acc10, false, false);
            acc11 = __builtin_amdgcn_wmma_f32_16x16x32_bf16(
                        false, a1.v, false, b1.v, (short)0, acc11, false, false);
        }
        const float en0 = enorm[kc + m];
        const float en1 = enorm[kc + 16 + m];
        const int  code0 = kc + m;
        const int  code1 = kc + 16 + m;
#pragma unroll
        for (int i = 0; i < 8; ++i) {
            // lower code chunk first, strict < : ties keep the smaller index
            float d00 = fmaf(-2.0f, acc00[i], en0);
            if (d00 < bestVal[0][i]) { bestVal[0][i] = d00; bestIdx[0][i] = code0; }
            float d01 = fmaf(-2.0f, acc01[i], en1);
            if (d01 < bestVal[0][i]) { bestVal[0][i] = d01; bestIdx[0][i] = code1; }
            float d10 = fmaf(-2.0f, acc10[i], en0);
            if (d10 < bestVal[1][i]) { bestVal[1][i] = d10; bestIdx[1][i] = code0; }
            float d11 = fmaf(-2.0f, acc11[i], en1);
            if (d11 < bestVal[1][i]) { bestVal[1][i] = d11; bestIdx[1][i] = code1; }
        }
    }

    // Reduce over the 16 code-lanes of each half; ties -> smaller index
#pragma unroll
    for (int t = 0; t < 2; ++t) {
#pragma unroll
        for (int i = 0; i < 8; ++i) {
            float v  = bestVal[t][i];
            int   ix = bestIdx[t][i];
#pragma unroll
            for (int off = 8; off >= 1; off >>= 1) {
                float ov  = __shfl_xor(v, off, 32);
                int   oix = __shfl_xor(ix, off, 32);
                if (ov < v || (ov == v && oix < ix)) { v = ov; ix = oix; }
            }
            if ((lane & 15) == 0) {
                unsigned key = __float_as_uint(v);
                key = (key & 0x80000000u) ? ~key : (key | 0x80000000u);  // order-preserving
                unsigned long long packed = ((unsigned long long)key << 32) | (unsigned)ix;
                atomicMin(&sBest[rhalf * 32 + t * 16 + hi * 8 + i], packed);  // ds_min_u64
            }
        }
    }
    __syncthreads();
    if (tid < 64) enc_idx[rowBase + tid] = (int)(sBest[tid] & 0xFFFFFFFFull);
}

// ---------------------------------------------------------------------------
// Gather quantized rows, loss partial sums, histogram, enc_idx as float
// ---------------------------------------------------------------------------
__global__ __launch_bounds__(256)
void gather_kernel(const float* __restrict__ x, const float* __restrict__ e,
                   const int* __restrict__ enc_idx, float* __restrict__ out,
                   float* __restrict__ lossAcc, unsigned* __restrict__ counts) {
    __shared__ float red[8];
    const int n = blockIdx.x, d = threadIdx.x;
    const int idx = enc_idx[n];
    float q  = e[(size_t)d * Kq + idx];
    float xv = x[(size_t)n * Dq + d];
    out[(size_t)n * Dq + d] = q;                    // quantized_st == quantized (value)
    float df = q - xv;
    float s  = df * df;
#pragma unroll
    for (int off = 16; off >= 1; off >>= 1) s += __shfl_xor(s, off, 32);
    if ((threadIdx.x & 31) == 0) red[threadIdx.x >> 5] = s;
    __syncthreads();
    if (threadIdx.x == 0) {
        float t = 0.f;
        for (int i = 0; i < 8; ++i) t += red[i];
        atomicAdd(lossAcc, t);
        atomicAdd(&counts[idx], 1u);
        out[(size_t)Nq * Dq + 2 + n] = (float)idx;  // enc_idx (4th output, as float)
    }
}

__global__ __launch_bounds__(256)
void final_kernel(const float* __restrict__ lossAcc,
                  const unsigned* __restrict__ counts, float* __restrict__ out) {
    __shared__ float red[256];
    float s = 0.f;
    for (int k = threadIdx.x; k < Kq; k += 256) {
        float p = (float)counts[k] * (1.0f / (float)Nq);
        s += p * logf(p + 1e-10f);
    }
    red[threadIdx.x] = s;
    __syncthreads();
    for (int st = 128; st >= 1; st >>= 1) {
        if (threadIdx.x < st) red[threadIdx.x] += red[threadIdx.x + st];
        __syncthreads();
    }
    if (threadIdx.x == 0) {
        out[(size_t)Nq * Dq]     = lossAcc[0] / (float)((size_t)Nq * Dq);  // loss
        out[(size_t)Nq * Dq + 1] = expf(-red[0]);                          // perplexity
    }
}

// ---------------------------------------------------------------------------
extern "C" void kernel_launch(void* const* d_in, const int* in_sizes, int n_in,
                              void* d_out, int out_size, void* d_ws, size_t ws_size,
                              hipStream_t stream) {
    (void)in_sizes; (void)n_in; (void)out_size; (void)ws_size;
    const float* x = (const float*)d_in[0];   // (B,T,D) -> (N, D)
    const float* e = (const float*)d_in[1];   // (D, K)
    float* out = (float*)d_out;

    char* ws = (char*)d_ws;
    size_t off = 0;
    __bf16* X2    = (__bf16*)(ws + off); off += (size_t)Nq * DP * 2;  // 24 MB
    __bf16* E2t   = (__bf16*)(ws + off); off += (size_t)Kq * DP * 2;  // 12 MB
    float*  enorm = (float*)(ws + off);  off += (size_t)Kq * 4;
    int*    enc   = (int*)(ws + off);    off += (size_t)Nq * 4;
    unsigned* counts = (unsigned*)(ws + off); off += (size_t)Kq * 4;
    float*  lossAcc  = (float*)(ws + off);

    // zero histogram + loss accumulator (memset node, capture-safe)
    hipMemsetAsync(counts, 0, (size_t)Kq * 4 + 16, stream);

    pack_x_kernel<<<(Nq * Dq) / 256, 256, 0, stream>>>(x, X2);
    pack_e_kernel<<<dim3(Kq / 256, Dq), 256, 0, stream>>>(e, E2t);
    enorm_kernel<<<Kq / 256, 256, 0, stream>>>(e, enorm);
    vq_argmin_kernel<<<Nq / 64, 256, 0, stream>>>(X2, E2t, enorm, enc);
    gather_kernel<<<Nq, 256, 0, stream>>>(x, e, enc, out, lossAcc, counts);
    final_kernel<<<1, 256, 0, stream>>>(lossAcc, counts, out);
}
